// iekt_10342281249182
// MI455X (gfx1250) — matrix operand order repaired and checked by reading, not compiled, not run on hardware
//
#include <hip/hip_runtime.h>
#include <math.h>

typedef __attribute__((ext_vector_type(16))) __bf16 v16bf;
typedef __attribute__((ext_vector_type(8)))  __bf16 v8bf;
typedef __attribute__((ext_vector_type(8)))  float  v8f;

#define DIM   128
#define K5    640     // 5*D
#define MBLK  32      // rows per block (2 x 16 M-subtiles)
#define LDA   648     // LDS row stride in bf16 elements (pad 8; 1296B, 16B aligned)

__device__ __forceinline__ __bf16 f2bf(float x) {
  union { float f; unsigned u; } a; a.f = x;
  unsigned r = a.u + 0x7FFFu + ((a.u >> 16) & 1u);
  union { unsigned short s; __bf16 b; } c; c.s = (unsigned short)(r >> 16);
  return c.b;
}

union U16 { v16bf v; v8bf h[2]; };

// ---------------- weight transpose/convert to bf16 ----------------
// WT[n*K + k] = W[k*N + n]
__global__ void k_transpose_w(const float* __restrict__ W, __bf16* __restrict__ WT,
                              int K, int N) {
  int idx = blockIdx.x * blockDim.x + threadIdx.x;
  if (idx >= K * N) return;
  int n = idx / K, k = idx % K;
  WT[(size_t)n * K + k] = f2bf(W[(size_t)k * N + n]);
}

// concat [Wr | Wz] (each 640x128) -> WT [256][640]
__global__ void k_transpose_wrz(const float* __restrict__ Wr, const float* __restrict__ Wz,
                                __bf16* __restrict__ WT) {
  int idx = blockIdx.x * blockDim.x + threadIdx.x;
  if (idx >= 256 * K5) return;
  int n = idx / K5, k = idx % K5;
  float v = (n < DIM) ? Wr[(size_t)k * DIM + n] : Wz[(size_t)k * DIM + (n - DIM)];
  WT[(size_t)n * K5 + k] = f2bf(v);
}

// ---------------- gathers + assemble X = [h, concept_rep, item, emb] (bf16) ----------------
__global__ void k_assemble(const float* __restrict__ h, const float* __restrict__ emb,
                           const int* __restrict__ pid, const int* __restrict__ rci,
                           const float* __restrict__ filt,
                           const float* __restrict__ cemb, const float* __restrict__ pemb,
                           const float* __restrict__ b2, float* __restrict__ prob_out,
                           __bf16* __restrict__ X) {
  int b = blockIdx.x;
  int t = threadIdx.x;           // 0..127 = feature dim
  float csum = 0.f, fsum = 0.f;
  #pragma unroll
  for (int m = 0; m < 8; ++m) {
    int idx = rci[b * 8 + m];
    if (idx != 0) csum += cemb[(size_t)(idx - 1) * DIM + t];
    fsum += filt[b * 8 + m];
  }
  float div = (fsum == 0.f) ? 1.f : fsum;
  float crep = csum / div;
  int p = pid[b];
  float item = (p == 0) ? 0.f : pemb[(size_t)(p - 1) * DIM + t];
  size_t r0 = (size_t)b * K5;
  X[r0 + t]        = f2bf(h[(size_t)b * DIM + t]);
  X[r0 + 128 + t]  = f2bf(crep);
  X[r0 + 256 + t]  = f2bf(item);
  X[r0 + 384 + t]  = f2bf(emb[(size_t)b * 2 * DIM + t]);
  X[r0 + 512 + t]  = f2bf(emb[(size_t)b * 2 * DIM + 128 + t]);
  if (t == 0) prob_out[b] = b2[0];   // prob accumulator seeded with bias2
}

// ---------------- fused WMMA GEMM ----------------
// MODE 0: pred   A=X,             W1T [640], epilogue relu*w2 -> atomicAdd prob
// MODE 1: r/z    A=[inputs,h],    WrzT [256], epilogue sigmoid -> RH(bf16), Z(f32)
// MODE 2: n-gate A=[inputs,r*h],  WnT  [128], epilogue tanh -> next_h
// All A-tiles are pure 16B-chunk gathers (operate is exactly 0/1, so the
// response blend is a row-uniform select) -> async global->LDS for everything.
template <int MODE>
__global__ __launch_bounds__(256) void k_gemm(
    const __bf16* __restrict__ X, const __bf16* __restrict__ WT,
    const float* __restrict__ bias0, const float* __restrict__ bias1,
    const float* __restrict__ aux0,   // MODE0: w2 ; MODE1/2: h
    const float* __restrict__ aux1,   // MODE2: Z
    const float* __restrict__ operate,
    __bf16* __restrict__ RH, float* __restrict__ Z,
    float* __restrict__ out, int B) {
  __shared__ __align__(16) __bf16 smA[MBLK * LDA];
  const int row0 = blockIdx.x * MBLK;

  // ---- build A tile in LDS (8 threads per row, 80 cols = 10 x 16B async copies) ----
  {
    const int r  = threadIdx.x >> 3;
    const int c0 = (threadIdx.x & 7) * 80;     // multiple of 8 elements (16B)
    const int gr = row0 + r;
    const __bf16* xr = X + (size_t)gr * K5;
    const unsigned ldsrow = (unsigned)(size_t)(void*)(smA + r * LDA);

    if (MODE == 0) {
      // straight copy; the instruction offset advances BOTH the LDS and the
      // global address (ISA §10 async pseudocode), so one base pair suffices.
      const __bf16* gsrc = xr + c0;
      const unsigned lds = ldsrow + c0 * 2;
      #pragma unroll
      for (int j = 0; j < 10; ++j) {
        asm volatile("global_load_async_to_lds_b128 %0, %1, off offset:%2"
                     :: "v"(lds), "v"(gsrc), "i"(j * 16) : "memory");
      }
    } else {
      // operate[gr] is exactly 0.0 or 1.0 (round of uniform): the blend
      // v*op + emb*(1-op) is an exact row-uniform select between X regions.
      const bool sel = (operate[gr] != 0.f);
      const __bf16* s1 = sel ? (xr + 128) : (xr + 384);   // inputs[0:256)
      const __bf16* s2 = sel ? (xr + 384) : (xr + 128);   // inputs[256:512)
      const __bf16* st = (MODE == 1) ? xr                  // tail = h (bf16 cols 0:128)
                                     : (RH + (size_t)gr * DIM);  // tail = r*h
      #pragma unroll
      for (int j = 0; j < 10; ++j) {
        const int c = c0 + j * 8;   // 8-elem chunks never straddle 256/512 boundaries
        const __bf16* g = (c < 256) ? (s1 + c)
                        : (c < 512) ? (s2 + (c - 256))
                                    : (st + (c - 512));
        asm volatile("global_load_async_to_lds_b128 %0, %1, off"
                     :: "v"(ldsrow + c * 2), "v"(g) : "memory");
      }
    }
    asm volatile("s_wait_asynccnt 0" ::: "memory");
  }
  __syncthreads();

  const int lane = threadIdx.x & 31;
  const int wave = threadIdx.x >> 5;
  const int l15  = lane & 15;
  const int hi   = lane >> 4;                     // 0: lanes 0-15, 1: lanes 16-31
  const int NT   = (MODE == 0) ? 40 : (MODE == 1 ? 16 : 8);

  float psum0[8], psum1[8];
  if (MODE == 0) {
    #pragma unroll
    for (int r = 0; r < 8; ++r) { psum0[r] = 0.f; psum1[r] = 0.f; }
  }

  const __bf16* arow0 = smA + (size_t)l15 * LDA + hi * 8;
  const __bf16* arow1 = smA + (size_t)(16 + l15) * LDA + hi * 8;

  for (int tn = wave; tn < NT; tn += 8) {
    const int ncol = tn * 16 + l15;
    const __bf16* wrow = WT + (size_t)ncol * K5 + hi * 8;
    v8f acc0 = {}; v8f acc1 = {};
    #pragma unroll 4
    for (int kt = 0; kt < 20; ++kt) {
      const int k0 = kt * 32;
      U16 bf_;
      bf_.h[0] = *(const v8bf*)(wrow + k0);
      bf_.h[1] = *(const v8bf*)(wrow + k0 + 16);
      U16 a0;
      a0.h[0] = *(const v8bf*)(arow0 + k0);
      a0.h[1] = *(const v8bf*)(arow0 + k0 + 16);
      U16 a1;
      a1.h[0] = *(const v8bf*)(arow1 + k0);
      a1.h[1] = *(const v8bf*)(arow1 + k0 + 16);
      acc0 = __builtin_amdgcn_wmma_f32_16x16x32_bf16(false, a0.v, false, bf_.v,
                                                     (short)0, acc0, false, false);
      acc1 = __builtin_amdgcn_wmma_f32_16x16x32_bf16(false, a1.v, false, bf_.v,
                                                     (short)0, acc1, false, false);
    }

    if (MODE == 0) {
      const float b1v = bias0[ncol];
      const float w2v = aux0[ncol];
      #pragma unroll
      for (int r = 0; r < 8; ++r) {
        float y0 = acc0[r] + b1v; y0 = y0 > 0.f ? y0 : 0.f; psum0[r] += y0 * w2v;
        float y1 = acc1[r] + b1v; y1 = y1 > 0.f ? y1 : 0.f; psum1[r] += y1 * w2v;
      }
    } else if (MODE == 1) {
      const float bv = (ncol < DIM) ? bias0[ncol] : bias1[ncol - DIM];
      #pragma unroll
      for (int r = 0; r < 8; ++r) {
        int ra = row0 + hi * 8 + r;
        int rb = row0 + 16 + hi * 8 + r;
        float s0 = 1.f / (1.f + __expf(-(acc0[r] + bv)));
        float s1 = 1.f / (1.f + __expf(-(acc1[r] + bv)));
        if (ncol < DIM) {   // r-gate -> r*h (bf16)
          RH[(size_t)ra * DIM + ncol] = f2bf(s0 * aux0[(size_t)ra * DIM + ncol]);
          RH[(size_t)rb * DIM + ncol] = f2bf(s1 * aux0[(size_t)rb * DIM + ncol]);
        } else {            // z-gate (f32)
          Z[(size_t)ra * DIM + (ncol - DIM)] = s0;
          Z[(size_t)rb * DIM + (ncol - DIM)] = s1;
        }
      }
    } else {
      const float bv = bias0[ncol];
      #pragma unroll
      for (int r = 0; r < 8; ++r) {
        int ra = row0 + hi * 8 + r;
        int rb = row0 + 16 + hi * 8 + r;
        float n0 = tanhf(acc0[r] + bv);
        float n1 = tanhf(acc1[r] + bv);
        float za = aux1[(size_t)ra * DIM + ncol], ha = aux0[(size_t)ra * DIM + ncol];
        float zb = aux1[(size_t)rb * DIM + ncol], hb = aux0[(size_t)rb * DIM + ncol];
        out[(size_t)B + (size_t)ra * DIM + ncol] = (1.f - za) * n0 + za * ha;
        out[(size_t)B + (size_t)rb * DIM + ncol] = (1.f - zb) * n1 + zb * hb;
      }
    }
  }

  if (MODE == 0) {
    // reduce over the 16 columns held by each half-wave (xor stays within half)
    #pragma unroll
    for (int off = 1; off < 16; off <<= 1) {
      #pragma unroll
      for (int r = 0; r < 8; ++r) {
        psum0[r] += __shfl_xor(psum0[r], off, 32);
        psum1[r] += __shfl_xor(psum1[r], off, 32);
      }
    }
    if (l15 == 0) {
      #pragma unroll
      for (int r = 0; r < 8; ++r) {
        atomicAdd(&out[row0 + hi * 8 + r], psum0[r]);
        atomicAdd(&out[row0 + 16 + hi * 8 + r], psum1[r]);
      }
    }
  }
}

extern "C" void kernel_launch(void* const* d_in, const int* in_sizes, int n_in,
                              void* d_out, int out_size, void* d_ws, size_t ws_size,
                              hipStream_t stream) {
  const float* h    = (const float*)d_in[0];
  const float* emb  = (const float*)d_in[1];
  const float* oper = (const float*)d_in[2];
  const int*   pid  = (const int*)d_in[3];
  const int*   rci  = (const int*)d_in[4];
  const float* filt = (const float*)d_in[5];
  const float* cemb = (const float*)d_in[6];
  const float* pemb = (const float*)d_in[7];
  const float* w1   = (const float*)d_in[8];
  const float* b1   = (const float*)d_in[9];
  const float* w2   = (const float*)d_in[10];
  const float* b2   = (const float*)d_in[11];
  const float* wr   = (const float*)d_in[12];
  const float* br   = (const float*)d_in[13];
  const float* wz   = (const float*)d_in[14];
  const float* bz   = (const float*)d_in[15];
  const float* wn   = (const float*)d_in[16];
  const float* bn   = (const float*)d_in[17];
  const int B = in_sizes[3];          // prob_ids has B elements
  float* out = (float*)d_out;

  // workspace carve-out (256B aligned)
  char* ws = (char*)d_ws;
  size_t off = 0;
  auto carve = [&](size_t bytes) -> void* {
    void* p = ws + off;
    off = (off + bytes + 255) & ~((size_t)255);
    return p;
  };
  __bf16* X    = (__bf16*)carve((size_t)B * K5 * 2);
  __bf16* RH   = (__bf16*)carve((size_t)B * DIM * 2);
  float*  Z    = (float*) carve((size_t)B * DIM * 4);
  __bf16* W1T  = (__bf16*)carve((size_t)K5 * K5 * 2);
  __bf16* WRZT = (__bf16*)carve((size_t)256 * K5 * 2);
  __bf16* WNT  = (__bf16*)carve((size_t)DIM * K5 * 2);
  (void)n_in; (void)out_size; (void)ws_size;

  k_transpose_w  <<<(K5 * K5 + 255) / 256, 256, 0, stream>>>(w1, W1T, K5, K5);
  k_transpose_wrz<<<(256 * K5 + 255) / 256, 256, 0, stream>>>(wr, wz, WRZT);
  k_transpose_w  <<<(DIM * K5 + 255) / 256, 256, 0, stream>>>(wn, WNT, K5, DIM);

  k_assemble<<<B, DIM, 0, stream>>>(h, emb, pid, rci, filt, cemb, pemb, b2, out, X);

  const int gb = B / MBLK;
  k_gemm<0><<<gb, 256, 0, stream>>>(X, W1T,  b1, nullptr, w2, nullptr, oper, RH, Z, out, B);
  k_gemm<1><<<gb, 256, 0, stream>>>(X, WRZT, br, bz,      h,  nullptr, oper, RH, Z, out, B);
  k_gemm<2><<<gb, 256, 0, stream>>>(X, WNT,  bn, nullptr, h,  Z,       oper, RH, Z, out, B);
}
